// DecoderGRU_37108517438022
// MI455X (gfx1250) — matrix operand order repaired and checked by reading, not compiled, not run on hardware
//
#include <hip/hip_runtime.h>

// ---------------------------------------------------------------------------
// DecoderGRU for MI455X (gfx1250): all matmuls via v_wmma_f32_16x16x32_bf16.
//
// Shapes: B=64, T=512, IN=256, H=512, G=512, 3H=1536.
// Plan:
//   (a) convert x, global_hidden and all weights fp32 -> bf16 (one pass)
//   (b) gm0/gm1 = g @ Wm[:, :G]^T + bm           (small WMMA GEMM)
//   (c) xt_all  = x @ Wx0^T + bx0                (big WMMA GEMM, 4 N-tiles/wave)
//   (d) scan over T: per step, 5 small kernels:
//         K1 : hx0 = gm0 + h0 @ Wm0h^T              (WMMA, f32+bf16 out)
//         K2 : fused Wh0 GEMM (r/u/n tiles) + GRU gate -> h0 (bf16)
//         K3a: xt1 = h0 @ Wx1^T + bx1               (WMMA, bf16 out)
//         K3b: hx1 = gm1 + h1 @ Wm1h^T              (WMMA, f32+bf16 out)
//         K4 : fused Wh1 GEMM + gate -> h1 (bf16) and out[:,t,:] (f32)
// Weights (~5.5 MB bf16) stay resident in the 192 MB L2 across the scan.
// ---------------------------------------------------------------------------

#define DEVINL __device__ __forceinline__

typedef __attribute__((ext_vector_type(16))) __bf16 v16bf;
typedef __attribute__((ext_vector_type(8)))  __bf16 v8bf;
typedef __attribute__((ext_vector_type(8)))  float  v8f;

namespace {
constexpr int kB  = 64;
constexpr int kT  = 512;
constexpr int kIN = 256;
constexpr int kH  = 512;
constexpr int kG  = 512;
constexpr int kH3 = 1536;
}

// ---- bf16 <-> f32 helpers (round-to-nearest-even) -------------------------
DEVINL __bf16 f2bf(float f) {
  unsigned u = __builtin_bit_cast(unsigned, f);
  unsigned r = u + 0x7FFFu + ((u >> 16) & 1u);
  unsigned short h = (unsigned short)(r >> 16);
  return __builtin_bit_cast(__bf16, h);
}
DEVINL float bf2f(__bf16 b) {
  unsigned short h = __builtin_bit_cast(unsigned short, b);
  unsigned u = ((unsigned)h) << 16;
  return __builtin_bit_cast(float, u);
}

// ---- WMMA fragment loads (contiguous-K layout, wave32) ----------------------
// A (MxK row-major, 16x32 tile): lane L -> row = L&15,
//   half h: K = 16*(L>>4) + h
DEVINL v16bf load_a_frag(const __bf16* __restrict__ A, int lda,
                         int mbase, int kbase, int lane) {
  const int r    = lane & 15;
  const int half = lane >> 4;
  const __bf16* p = A + (size_t)(mbase + r) * lda + kbase + half * 16;
  union { v16bf v; v8bf h[2]; } u;
  u.h[0] = *(const v8bf*)(p);        // K = half*16 .. +7
  u.h[1] = *(const v8bf*)(p + 8);    // K = half*16 + 8 .. +15
  return u.v;
}
// B = W^T where W is (N x K) row-major. 32x16 tile: lane L -> col = L&15,
//   half h: K = 16*(L>>4) + h  -> 16 contiguous bf16 of one W row.
DEVINL v16bf load_b_frag(const __bf16* __restrict__ W, int ldw,
                         int nbase, int kbase, int lane) {
  const int n    = lane & 15;
  const int koff = (lane >> 4) * 16;
  const __bf16* p = W + (size_t)(nbase + n) * ldw + kbase + koff;
  union { v16bf v; v8bf h[2]; } u;
  u.h[0] = *(const v8bf*)(p);
  u.h[1] = *(const v8bf*)(p + 8);
  return u.v;
}
DEVINL v8f wmma_bf16(v16bf a, v16bf b, v8f c) {
  return __builtin_amdgcn_wmma_f32_16x16x32_bf16(
      false, a, false, b, (short)0, c, false, false);
}

// ---- utility kernels -------------------------------------------------------
__global__ void f32_to_bf16_kernel(const float* __restrict__ in,
                                   __bf16* __restrict__ out, int n) {
  int i = blockIdx.x * blockDim.x + threadIdx.x;
  if (i < n) out[i] = f2bf(in[i]);
}
__global__ void zero_bf16_kernel(__bf16* __restrict__ p, int n) {
  int i = blockIdx.x * blockDim.x + threadIdx.x;
  if (i < n) p[i] = __builtin_bit_cast(__bf16, (unsigned short)0);
}

// ---- generic WMMA GEMM: C = A(MxK) @ W(NxK)^T [+ bias[n]] [+ addsrc[m,n]] --
// NT = N-tiles per wave (A-fragment reuse). Outputs optional f32 and/or bf16.
template <int NT>
__global__ void __launch_bounds__(128)
gemm_wmma_kernel(const __bf16* __restrict__ A, int lda,
                 const __bf16* __restrict__ W, int ldw,
                 const float* __restrict__ bias,
                 const float* __restrict__ addsrc, int ldadd,
                 float* __restrict__ Cf, int ldcf,
                 __bf16* __restrict__ Cb, int ldcb,
                 int M, int N, int K) {
  const int lane = threadIdx.x & 31;
  const int wave = threadIdx.x >> 5;
  const int w    = blockIdx.x * 4 + wave;
  const int tilesNg = (N >> 4) / NT;
  if (w >= (M >> 4) * tilesNg) return;
  const int mbase  = (w / tilesNg) << 4;
  const int nbase0 = (w % tilesNg) * (NT * 16);

  v8f acc[NT];
#pragma unroll
  for (int j = 0; j < NT; ++j) acc[j] = (v8f){};

  for (int kb = 0; kb < K; kb += 32) {
    v16bf a = load_a_frag(A, lda, mbase, kb, lane);
    if (kb + 32 < K)  // warm next K-block of B into cache (global_prefetch_b8)
      __builtin_prefetch(W + (size_t)(nbase0 + (lane & 15)) * ldw + kb + 32, 0, 3);
#pragma unroll
    for (int j = 0; j < NT; ++j) {
      v16bf b = load_b_frag(W, ldw, nbase0 + j * 16, kb, lane);
      acc[j] = wmma_bf16(a, b, acc[j]);
    }
  }

  const int nl = lane & 15;
  const int mo = (lane >> 4) << 3;
#pragma unroll
  for (int j = 0; j < NT; ++j) {
    const int n  = nbase0 + j * 16 + nl;
    const float bv = bias ? bias[n] : 0.f;
#pragma unroll
    for (int v = 0; v < 8; ++v) {
      const int m = mbase + mo + v;
      float val = acc[j][v] + bv;
      if (addsrc) val += addsrc[(size_t)m * ldadd + n];
      if (Cf) Cf[(size_t)m * ldcf + n] = val;
      if (Cb) Cb[(size_t)m * ldcb + n] = f2bf(val);
    }
  }
}

// ---- fused Wh GEMM + GRU gate ---------------------------------------------
// ht = hx @ Wh^T + bh  (r/u/n tiles share one A fragment)
// r = sig(xr+hr); u = sig(xu+hu); n = tanh(xn + r*hn); h = u*hx + (1-u)*n
__global__ void __launch_bounds__(128)
gru_gate_kernel(const __bf16* __restrict__ hx_bf,   // M x H
                const float*  __restrict__ hx_f,    // M x H
                const __bf16* __restrict__ Wh,      // 3H x H
                const float*  __restrict__ bh,      // 3H
                const __bf16* __restrict__ xt, int ldxt,  // M x 3H
                __bf16* __restrict__ hout_bf,       // M x H
                float*  __restrict__ hout_f, int ldhf,    // optional
                int M, int H) {
  const int lane = threadIdx.x & 31;
  const int wave = threadIdx.x >> 5;
  const int w    = blockIdx.x * 4 + wave;
  const int tilesN = H >> 4;
  if (w >= (M >> 4) * tilesN) return;
  const int mbase = (w / tilesN) << 4;
  const int nbase = (w % tilesN) << 4;

  v8f ar = (v8f){}, au = (v8f){}, an = (v8f){};
  for (int kb = 0; kb < H; kb += 32) {
    v16bf a  = load_a_frag(hx_bf, H, mbase, kb, lane);
    v16bf br = load_b_frag(Wh, H, nbase,         kb, lane);
    v16bf bu = load_b_frag(Wh, H, nbase + H,     kb, lane);
    v16bf bn = load_b_frag(Wh, H, nbase + 2 * H, kb, lane);
    ar = wmma_bf16(a, br, ar);
    au = wmma_bf16(a, bu, au);
    an = wmma_bf16(a, bn, an);
  }

  const int n  = nbase + (lane & 15);
  const int mo = (lane >> 4) << 3;
  const float bhr = bh[n], bhu = bh[n + H], bhn = bh[n + 2 * H];
#pragma unroll
  for (int v = 0; v < 8; ++v) {
    const int m = mbase + mo + v;
    const __bf16* xrow = xt + (size_t)m * ldxt;
    const float xr = bf2f(xrow[n]);
    const float xu = bf2f(xrow[n + H]);
    const float xn = bf2f(xrow[n + 2 * H]);
    const float r  = 1.f / (1.f + __expf(-(xr + ar[v] + bhr)));
    const float u  = 1.f / (1.f + __expf(-(xu + au[v] + bhu)));
    const float nn = tanhf(xn + r * (an[v] + bhn));
    const float hx = hx_f[(size_t)m * H + n];
    const float h  = u * hx + (1.f - u) * nn;
    hout_bf[(size_t)m * H + n] = f2bf(h);
    if (hout_f) hout_f[(size_t)m * ldhf + n] = h;
  }
}

// ---------------------------------------------------------------------------
extern "C" void kernel_launch(void* const* d_in, const int* in_sizes, int n_in,
                              void* d_out, int out_size, void* d_ws, size_t ws_size,
                              hipStream_t stream) {
  (void)in_sizes; (void)n_in; (void)out_size; (void)ws_size;
  const float* x   = (const float*)d_in[0];
  const float* g   = (const float*)d_in[1];
  const float* Wx0 = (const float*)d_in[2];
  const float* bx0 = (const float*)d_in[3];
  const float* Wh0 = (const float*)d_in[4];
  const float* bh0 = (const float*)d_in[5];
  const float* Wm0 = (const float*)d_in[6];
  const float* bm0 = (const float*)d_in[7];
  const float* Wx1 = (const float*)d_in[8];
  const float* bx1 = (const float*)d_in[9];
  const float* Wh1 = (const float*)d_in[10];
  const float* bh1 = (const float*)d_in[11];
  const float* Wm1 = (const float*)d_in[12];
  const float* bm1 = (const float*)d_in[13];
  float* out = (float*)d_out;

  // ---- carve workspace (256B aligned) ----
  char* p = (char*)d_ws;
  auto alloc = [&](size_t bytes) -> char* {
    char* r = p; p += (bytes + 255) & ~(size_t)255; return r;
  };
  __bf16* x_bf     = (__bf16*)alloc((size_t)kB * kT * kIN * 2);
  __bf16* g_bf     = (__bf16*)alloc((size_t)kB * kG * 2);
  __bf16* Wx0_bf   = (__bf16*)alloc((size_t)kH3 * kIN * 2);
  __bf16* Wh0_bf   = (__bf16*)alloc((size_t)kH3 * kH * 2);
  __bf16* Wm0_bf   = (__bf16*)alloc((size_t)kH * (kG + kH) * 2);
  __bf16* Wx1_bf   = (__bf16*)alloc((size_t)kH3 * kH * 2);
  __bf16* Wh1_bf   = (__bf16*)alloc((size_t)kH3 * kH * 2);
  __bf16* Wm1_bf   = (__bf16*)alloc((size_t)kH * (kG + kH) * 2);
  __bf16* xt_all   = (__bf16*)alloc((size_t)kB * kT * kH3 * 2);
  float*  gm0_f    = (float*)alloc((size_t)kB * kH * 4);
  float*  gm1_f    = (float*)alloc((size_t)kB * kH * 4);
  float*  hx0_f    = (float*)alloc((size_t)kB * kH * 4);
  float*  hx1_f    = (float*)alloc((size_t)kB * kH * 4);
  __bf16* hx0_bf   = (__bf16*)alloc((size_t)kB * kH * 2);
  __bf16* hx1_bf   = (__bf16*)alloc((size_t)kB * kH * 2);
  __bf16* h0_bf    = (__bf16*)alloc((size_t)kB * kH * 2);
  __bf16* h1_bf    = (__bf16*)alloc((size_t)kB * kH * 2);
  __bf16* xt1_bf   = (__bf16*)alloc((size_t)kB * kH3 * 2);

  auto cvt = [&](const float* src, __bf16* dst, int n) {
    f32_to_bf16_kernel<<<(n + 255) / 256, 256, 0, stream>>>(src, dst, n);
  };
  cvt(x,   x_bf,   kB * kT * kIN);
  cvt(g,   g_bf,   kB * kG);
  cvt(Wx0, Wx0_bf, kH3 * kIN);
  cvt(Wh0, Wh0_bf, kH3 * kH);
  cvt(Wm0, Wm0_bf, kH * (kG + kH));
  cvt(Wx1, Wx1_bf, kH3 * kH);
  cvt(Wh1, Wh1_bf, kH3 * kH);
  cvt(Wm1, Wm1_bf, kH * (kG + kH));

  const int nH = kB * kH;
  zero_bf16_kernel<<<(nH + 255) / 256, 256, 0, stream>>>(h0_bf, nH);
  zero_bf16_kernel<<<(nH + 255) / 256, 256, 0, stream>>>(h1_bf, nH);

  auto gemm = [&](int NT, const __bf16* A, int lda, const __bf16* W, int ldw,
                  const float* bias, const float* addsrc, int ldadd,
                  float* Cf, int ldcf, __bf16* Cb, int ldcb,
                  int M, int N, int K) {
    const int groups = (M / 16) * ((N / 16) / NT);
    const int blocks = (groups + 3) / 4;
    if (NT == 4)
      gemm_wmma_kernel<4><<<blocks, 128, 0, stream>>>(
          A, lda, W, ldw, bias, addsrc, ldadd, Cf, ldcf, Cb, ldcb, M, N, K);
    else
      gemm_wmma_kernel<1><<<blocks, 128, 0, stream>>>(
          A, lda, W, ldw, bias, addsrc, ldadd, Cf, ldcf, Cb, ldcb, M, N, K);
  };

  // gm0 = g @ Wm0[:, :G]^T + bm0 ; gm1 likewise
  gemm(1, g_bf, kG, Wm0_bf, kG + kH, bm0, nullptr, 0,
       gm0_f, kH, nullptr, 0, kB, kH, kG);
  gemm(1, g_bf, kG, Wm1_bf, kG + kH, bm1, nullptr, 0,
       gm1_f, kH, nullptr, 0, kB, kH, kG);

  // xt_all = x @ Wx0^T + bx0   (big GEMM, 4 N-tiles per wave)
  gemm(4, x_bf, kIN, Wx0_bf, kIN, bx0, nullptr, 0,
       nullptr, 0, xt_all, kH3, kB * kT, kH3, kIN);

  const int gateBlocks = ((kB / 16) * (kH / 16) + 3) / 4;  // 32
  for (int t = 0; t < kT; ++t) {
    // K1: hx0 = gm0 + h0 @ Wm0h^T
    gemm(1, h0_bf, kH, Wm0_bf + kG, kG + kH, nullptr, gm0_f, kH,
         hx0_f, kH, hx0_bf, kH, kB, kH, kH);
    // K2: fused Wh0 GEMM + gates -> h0
    gru_gate_kernel<<<gateBlocks, 128, 0, stream>>>(
        hx0_bf, hx0_f, Wh0_bf, bh0,
        xt_all + (size_t)t * kH3, kT * kH3,
        h0_bf, nullptr, 0, kB, kH);
    // K3a: xt1 = h0 @ Wx1^T + bx1
    gemm(1, h0_bf, kH, Wx1_bf, kH, bx1, nullptr, 0,
         nullptr, 0, xt1_bf, kH3, kB, kH3, kH);
    // K3b: hx1 = gm1 + h1 @ Wm1h^T
    gemm(1, h1_bf, kH, Wm1_bf + kG, kG + kH, nullptr, gm1_f, kH,
         hx1_f, kH, hx1_bf, kH, kB, kH, kH);
    // K4: fused Wh1 GEMM + gates -> h1, and out[:, t, :]
    gru_gate_kernel<<<gateBlocks, 128, 0, stream>>>(
        hx1_bf, hx1_f, Wh1_bf, bh1,
        xt1_bf, kH3,
        h1_bf, out + (size_t)t * kH, kT * kH, kB, kH);
  }
}
